// IterativeRefiner_87686052315369
// MI455X (gfx1250) — matrix operand order repaired and checked by reading, not compiled, not run on hardware
//
#include <hip/hip_runtime.h>
#include <hip/hip_bf16.h>
#include <math.h>

// ---------------------------------------------------------------------------
// Model constants (from the reference)
// ---------------------------------------------------------------------------
constexpr int Bc  = 8;    // batch
constexpr int NV  = 384;  // nodes
constexpr int NE  = 96;   // edges
constexpr int Tit = 3;    // refine iterations

typedef __attribute__((ext_vector_type(8)))  float          f32x8;
typedef __attribute__((ext_vector_type(8)))  unsigned short u16x8;
typedef __attribute__((ext_vector_type(16))) __bf16         bf16x16;

struct U16x16 { u16x8 lo, hi; };

// ---------------------------------------------------------------------------
// Device helpers
// ---------------------------------------------------------------------------
__device__ __forceinline__ unsigned short f2bf(float x) {
  unsigned int u = __builtin_bit_cast(unsigned int, x);
  u += 0x7fffu + ((u >> 16) & 1u);           // round-to-nearest-even
  return (unsigned short)(u >> 16);
}
__device__ __forceinline__ float bf2f(unsigned short h) {
  unsigned int u = ((unsigned int)h) << 16;
  return __builtin_bit_cast(float, u);
}
__device__ __forceinline__ float geluf(float x) {  // tanh approx (jax default)
  float x3 = x * x * x;
  return 0.5f * x * (1.f + tanhf(0.7978845608028654f * (x + 0.044715f * x3)));
}
__device__ __forceinline__ float waveSum(float v) {
  for (int o = 16; o > 0; o >>= 1) v += __shfl_xor(v, o, 32);
  return v;
}
__device__ __forceinline__ float waveMax(float v) {
  for (int o = 16; o > 0; o >>= 1) v = fmaxf(v, __shfl_xor(v, o, 32));
  return v;
}
__device__ float blockSum256(float v) {  // blockDim.x == 256
  __shared__ float sm[8];
  int lane = threadIdx.x & 31, wid = threadIdx.x >> 5;
  v = waveSum(v);
  if (lane == 0) sm[wid] = v;
  __syncthreads();
  if (wid == 0) {
    float r = (lane < 8) ? sm[lane] : 0.f;
    r = waveSum(r);
    if (lane == 0) sm[0] = r;
  }
  __syncthreads();
  float r = sm[0];
  __syncthreads();
  return r;
}
// two 16-byte aligned contiguous chunks -> one bf16x16 WMMA fragment
__device__ __forceinline__ bf16x16 ldfrag(const unsigned short* p0,
                                          const unsigned short* p1) {
  U16x16 t;
  t.lo = *(const u16x8*)p0;
  t.hi = *(const u16x8*)p1;
  return __builtin_bit_cast(bf16x16, t);
}

// ---------------------------------------------------------------------------
// bf16 WMMA GEMM with K-contiguous (transposed) B:
//   C[M,N](f32) = act(alpha * A[M,K] @ Bt[N,K]^T + bias - mvec) (+gated residual)
// One wave computes a 32x32 tile (4 accumulators). K % 32 == 0, M % 32 == 0,
// N % 32 == 0. Fragment layouts per CDNA5 ISA 7.12.2:
//   A lane(row m=m0+row[,+16], half): elems j=0..7 -> k=half*8+j, j=8..15 ->
//     k=16+half*8+j-8  => two contiguous 8-elem (16B) runs.
//   B lane(col n=n0+row[,+16], half): same k mapping as A => two 8-elem runs.
// All chunk addresses are 16B-aligned (every lda/ldbt used is a multiple of 16).
// ---------------------------------------------------------------------------
__global__ __launch_bounds__(32)
void gemm_bf16(const unsigned short* __restrict__ A, int lda, long sA,
               const unsigned short* __restrict__ Bt, int ldbt, long sB,
               float* __restrict__ C, int ldc, long sC,
               unsigned short* __restrict__ Cb, int ldcb, long sCb,
               int K,
               const float* __restrict__ bias,
               const float* __restrict__ res, int ldres, long sres,
               const float* __restrict__ gate, int gst,
               const float* __restrict__ mvec, int mst,
               int rows_per_b, float alpha, int act) {
  const int lane = threadIdx.x;
  const int row = lane & 15, half = lane >> 4;
  const int n0 = blockIdx.x * 32, m0 = blockIdx.y * 32, z = blockIdx.z;
  const unsigned short* __restrict__ a0 = A + (long)z * sA + (long)(m0 + row) * lda;
  const unsigned short* __restrict__ a1 = a0 + (long)16 * lda;
  const unsigned short* __restrict__ b0 = Bt + (long)z * sB + (long)(n0 + row) * ldbt;
  const unsigned short* __restrict__ b1 = b0 + (long)16 * ldbt;
  f32x8 acc00 = {0.f, 0.f, 0.f, 0.f, 0.f, 0.f, 0.f, 0.f};
  f32x8 acc01 = acc00, acc10 = acc00, acc11 = acc00;
  const int ha = half * 8, hb = half * 16;
  for (int k0 = 0; k0 < K; k0 += 32) {
    bf16x16 af0 = ldfrag(a0 + k0 + ha, a0 + k0 + 16 + ha);
    bf16x16 af1 = ldfrag(a1 + k0 + ha, a1 + k0 + 16 + ha);
    bf16x16 bf0 = ldfrag(b0 + k0 + ha, b0 + k0 + 16 + ha);
    bf16x16 bf1 = ldfrag(b1 + k0 + ha, b1 + k0 + 16 + ha);
    acc00 = __builtin_amdgcn_wmma_f32_16x16x32_bf16(false, af0, false, bf0, (short)0, acc00, false, false);
    acc01 = __builtin_amdgcn_wmma_f32_16x16x32_bf16(false, af0, false, bf1, (short)0, acc01, false, false);
    acc10 = __builtin_amdgcn_wmma_f32_16x16x32_bf16(false, af1, false, bf0, (short)0, acc10, false, false);
    acc11 = __builtin_amdgcn_wmma_f32_16x16x32_bf16(false, af1, false, bf1, (short)0, acc11, false, false);
  }
  // tiles never straddle a batch-row boundary (rows_per_b % 32 == 0)
  const int bbc = (rows_per_b > 0) ? (m0 / rows_per_b) : z;
#pragma unroll
  for (int mi = 0; mi < 2; ++mi) {
#pragma unroll
    for (int ni = 0; ni < 2; ++ni) {
      f32x8 acc = mi ? (ni ? acc11 : acc10) : (ni ? acc01 : acc00);
      int n = n0 + ni * 16 + row;
      float bval = bias ? bias[n] : 0.f;
      float mval = mvec ? mvec[bbc * mst + n] : 0.f;
      float gval = gate ? gate[bbc * gst + n] : 1.f;
#pragma unroll
      for (int i = 0; i < 8; ++i) {
        int m = m0 + mi * 16 + half * 8 + i;
        float v = acc[i] * alpha + bval - mval;
        if (act == 1)      v = fmaxf(v, 0.f);
        else if (act == 2) v = geluf(v);
        if (res) v = res[(long)z * sres + (long)m * ldres + n] + gval * v;
        if (C)  C[(long)z * sC + (long)m * ldc + n] = v;
        if (Cb) Cb[(long)z * sCb + (long)m * ldcb + n] = f2bf(v);
      }
    }
  }
}

// ---------------------------------------------------------------------------
// Staging kernels (weights -> bf16, transposed so B is K-contiguous)
// ---------------------------------------------------------------------------
__global__ void cvtT_bf16_k(const float* s, int K, int N, unsigned short* d, int total) {
  int i = blockIdx.x * 256 + threadIdx.x;
  if (i < total) { int n = i / K, k = i % K; d[i] = f2bf(s[(long)k * N + n]); }
}
__global__ void add2T_bf16_k(const float* a, const float* b, int K, int N,
                             unsigned short* d, int total) {
  int i = blockIdx.x * 256 + threadIdx.x;
  if (i < total) { int n = i / K, k = i % K; long s = (long)k * N + n; d[i] = f2bf(a[s] + b[s]); }
}
__global__ void padT_proj_k(const float* s, unsigned short* d) {  // [16,256]->[256,32]
  int i = blockIdx.x * 256 + threadIdx.x;   // exact: 8192
  int n = i / 32, k = i % 32;
  d[i] = (k < 16) ? f2bf(s[k * 256 + n]) : (unsigned short)0;
}
__global__ void pad_cols_bf16_k(const float* s, int scols, int dcols, int rows,
                                unsigned short* d) {
  int i = blockIdx.x * 256 + threadIdx.x;
  if (i < rows * dcols) {
    int r = i / dcols, c = i % dcols;
    d[i] = (c < scols) ? f2bf(s[(long)r * scols + c]) : (unsigned short)0;
  }
}
__global__ void addf_k(const float* a, const float* b, float* d, int n) {
  int i = blockIdx.x * 256 + threadIdx.x;
  if (i < n) d[i] = a[i] + b[i];
}
// src [B*rpb, 256] f32 -> dst [B][256][rpb] bf16 (K-contiguous over rows)
__global__ void bf16T_k(const float* src, int rpb, unsigned short* dst) {
  int i = blockIdx.x * 256 + threadIdx.x;
  int rowi = i >> 8, t = i & 255;
  int bb = rowi / rpb, rr = rowi % rpb;
  dst[((long)bb * 256 + t) * rpb + rr] = f2bf(src[i]);
}
__global__ void silu_k(const float* s, float* d, int n) {
  int i = blockIdx.x * 256 + threadIdx.x;
  if (i < n) { float x = s[i]; d[i] = x / (1.f + expf(-x)); }
}
// e_t f32 -> bf16 into first 256 cols of the 512-wide concat buffer
__global__ void pack_e_k(const float* e, unsigned short* cat2) {
  int i = blockIdx.x * 256 + threadIdx.x;   // exact: 196608
  int r = i / 256, c = i & 255;
  cat2[(long)r * 512 + c] = f2bf(e[i]);
}
__global__ void e_init_k(const float* mu, const float* ls, const float* nz, float* e) {
  int i = blockIdx.x * 256 + threadIdx.x;   // exact: 196608
  int d = i & 255;
  e[i] = mu[d] + expf(ls[d]) * nz[i];
}
__global__ void i_init_k(const int* mask, float* it) {
  int i = blockIdx.x * 256 + threadIdx.x;   // exact: 294912
  int b = i / (NE * NV); int r = i % (NE * NV); int e = r / NV; int n = r % NV;
  float mf = mask[b * NV + n] ? 1.f : 0.f;
  it[i] = (1.f / (float)NE) * (1.f - mf) + ((e == n) ? mf : 0.f);
}

// ---------------------------------------------------------------------------
// Fused OutCatLinear: ilog[b,e,n] = sum_d relu(pn[b,n,d]+pe[b,e,d]+i*piw+pib)*ow + ob
// grid(NV/8, NE, B), block 256 (8 waves, wave -> one n)
// ---------------------------------------------------------------------------
__global__ __launch_bounds__(256)
void outcat_k(const float* __restrict__ pn, const float* __restrict__ pe,
              const float* __restrict__ it, const float* __restrict__ piw,
              const float* __restrict__ pib, const float* __restrict__ ow,
              const float* __restrict__ ob, float* __restrict__ ilog) {
  __shared__ float peS[256], piwS[256], pibS[256], owS[256];
  int t = threadIdx.x, e = blockIdx.y, b = blockIdx.z;
  peS[t]  = pe[((long)(b * NE + e)) * 256 + t];
  piwS[t] = piw[t]; pibS[t] = pib[t]; owS[t] = ow[t];
  __syncthreads();
  int wid = t >> 5, lane = t & 31;
  int n = blockIdx.x * 8 + wid;
  float iv = it[((long)(b * NE + e)) * NV + n];
  const float* pr = pn + ((long)(b * NV + n)) * 256;
  float s = 0.f;
#pragma unroll
  for (int q = 0; q < 8; ++q) {
    int d = lane + 32 * q;
    float tv = pr[d] + peS[d] + iv * piwS[d] + pibS[d];
    s += fmaxf(tv, 0.f) * owS[d];
  }
  s = waveSum(s);
  if (lane == 0) ilog[((long)(b * NE + e)) * NV + n] = s + ob[0];
}

// softmax over e (96) per (b,n), then mask/teye; wave per column
__global__ __launch_bounds__(128)
void softmax_col_k(const float* __restrict__ ilog, const int* __restrict__ mask,
                   float* __restrict__ it) {
  int wid = threadIdx.x >> 5, lane = threadIdx.x & 31;
  int col = blockIdx.x * 4 + wid;           // 0..3071
  int b = col / NV, n = col % NV;
  const float* base = ilog + (long)b * NE * NV + n;
  float v0 = base[(long)lane * NV];
  float v1 = base[(long)(lane + 32) * NV];
  float v2 = base[(long)(lane + 64) * NV];
  float mx = waveMax(fmaxf(v0, fmaxf(v1, v2)));
  v0 = expf(v0 - mx); v1 = expf(v1 - mx); v2 = expf(v2 - mx);
  float inv = 1.f / waveSum(v0 + v1 + v2);
  float mf = mask[b * NV + n] ? 1.f : 0.f;
  float keep = 1.f - mf;
  float* o = it + (long)b * NE * NV + n;
  o[(long)lane * NV]        = v0 * inv * keep + ((lane == n)      ? mf : 0.f);
  o[(long)(lane + 32) * NV] = v1 * inv * keep + ((lane + 32 == n) ? mf : 0.f);
  o[(long)(lane + 64) * NV] = v2 * inv * keep + ((lane + 64 == n) ? mf : 0.f);
}

__global__ __launch_bounds__(256)
void isum_k(const float* __restrict__ it, float* __restrict__ isum) {
  int wid = threadIdx.x >> 5, lane = threadIdx.x & 31;
  int be = blockIdx.x * 8 + wid;            // 0..767
  const float* r = it + (long)be * NV;
  float s = 0.f;
#pragma unroll
  for (int q = 0; q < 12; ++q) s += r[lane + 32 * q];
  s = waveSum(s);
  if (lane == 0) isum[be] = s;
}

// edge-indicator MLP: relu([e_t,i_sum]@w1+b1)@w2+b2 -> mask -> sigmoid
__global__ __launch_bounds__(256)
void edge_ind_k(const float* __restrict__ et, const float* __restrict__ isum,
                const float* __restrict__ w1, const float* __restrict__ b1,
                const float* __restrict__ w2, const float* __restrict__ b2,
                const int* __restrict__ mask, float* __restrict__ eind,
                float* __restrict__ logitb) {
  __shared__ float hsh[257];
  int be = blockIdx.x, t = threadIdx.x;
  int b = be / NE, e = be % NE;
  hsh[t] = et[(long)be * 256 + t];
  if (t == 0) hsh[256] = isum[be];
  __syncthreads();
  float acc = b1[t];
  for (int k = 0; k < 257; ++k) acc += hsh[k] * w1[k * 256 + t];
  acc = fmaxf(acc, 0.f);
  float tot = blockSum256(acc * w2[t]);
  if (t == 0) {
    float lg = tot + b2[0];
    if (mask[b * NV + e]) lg = 1e6f;
    logitb[be] = lg;
    eind[be] = 1.f / (1.f + expf(-lg));
  }
}

// im = i_new * e_ind  (bf16, plus transposed copy for upd_n GEMM)
__global__ void im_k(const float* it, const float* eind,
                     unsigned short* im, unsigned short* imT) {
  int i = blockIdx.x * 256 + threadIdx.x;   // exact: 294912
  int b = i / (NE * NV); int r = i % (NE * NV); int e = r / NV; int n = r % NV;
  unsigned short u = f2bf(it[i] * eind[b * NE + e]);
  im[i] = u;
  imT[((long)(b * NV + n)) * NE + e] = u;
}

// qkv [768,768] f32 -> q[z,96,32], k[z,96,32] (Bt for scores), vT[z,32,96]
// (Bt for PV), z = b*8+h
__global__ void qkv_split_k(const float* qkv, unsigned short* q,
                            unsigned short* kB, unsigned short* vT) {
  int i = blockIdx.x * 256 + threadIdx.x;   // exact: 589824
  int c = i % 768; int e = (i / 768) % NE; int b = i / (768 * NE);
  int which = c >> 8, h = (c & 255) >> 5, dh = c & 31;
  int z = b * 8 + h;
  unsigned short u = f2bf(qkv[i]);
  if (which == 0)      q[((z * NE + e) * 32) + dh] = u;
  else if (which == 1) kB[((z * NE + e) * 32) + dh] = u;
  else                 vT[((z * 32 + dh) * NE) + e] = u;
}

__global__ __launch_bounds__(128)
void softmax_row_k(const float* sc, unsigned short* a) {
  int wid = threadIdx.x >> 5, lane = threadIdx.x & 31;
  int rowi = blockIdx.x * 4 + wid;          // < 6144
  const float* r = sc + (long)rowi * NE;
  float v0 = r[lane], v1 = r[lane + 32], v2 = r[lane + 64];
  float mx = waveMax(fmaxf(v0, fmaxf(v1, v2)));
  v0 = expf(v0 - mx); v1 = expf(v1 - mx); v2 = expf(v2 - mx);
  float inv = 1.f / waveSum(v0 + v1 + v2);
  unsigned short* o = a + (long)rowi * NE;
  o[lane] = f2bf(v0 * inv); o[lane + 32] = f2bf(v1 * inv); o[lane + 64] = f2bf(v2 * inv);
}

__global__ void o_reorder_k(const unsigned short* ob, unsigned short* oc) {
  int i = blockIdx.x * 256 + threadIdx.x;   // exact: 196608
  int z = i / 3072; int rem = i % 3072; int e = rem / 32; int dh = rem & 31;
  int b = z / 8, h = z % 8;
  oc[((b * NE + e) * 256) + h * 32 + dh] = ob[i];
}

// ---------------------------------------------------------------------------
// LayerNorm family (row per block, blockDim 256)
// ---------------------------------------------------------------------------
__global__ __launch_bounds__(256)
void ln_mod256_k(const float* __restrict__ x, const float* __restrict__ sh,
                 const float* __restrict__ sc, int cstride, int rows_per_b,
                 unsigned short* __restrict__ out) {
  int rowi = blockIdx.x, t = threadIdx.x;
  float v = x[(long)rowi * 256 + t];
  float mean = blockSum256(v) * (1.f / 256.f);
  float d = v - mean;
  float var = blockSum256(d * d) * (1.f / 256.f);
  float nv = d * rsqrtf(var + 1e-5f);
  int bb = rowi / rows_per_b;
  out[(long)rowi * 256 + t] = f2bf(nv * (1.f + sc[bb * cstride + t]) + sh[bb * cstride + t]);
}
// optional outBT: [B][256][rpb_t] bf16 K-contiguous transposed copy
__global__ __launch_bounds__(256)
void ln_affine256_k(const float* __restrict__ s1, const float* __restrict__ s2,
                    const float* __restrict__ g, const float* __restrict__ bvec,
                    float* __restrict__ outF, unsigned short* __restrict__ outB,
                    int ldb_out, unsigned short* __restrict__ outBT, int rpb_t) {
  int rowi = blockIdx.x, t = threadIdx.x;
  float v = s1[(long)rowi * 256 + t];
  if (s2) v += s2[(long)rowi * 256 + t];
  float mean = blockSum256(v) * (1.f / 256.f);
  float d = v - mean;
  float var = blockSum256(d * d) * (1.f / 256.f);
  float o = d * rsqrtf(var + 1e-5f) * g[t] + bvec[t];
  if (outF) outF[(long)rowi * 256 + t] = o;
  if (outB) outB[(long)rowi * ldb_out + t] = f2bf(o);
  if (outBT) {
    int bb = rowi / rpb_t, rr = rowi % rpb_t;
    outBT[((long)bb * 256 + t) * rpb_t + rr] = f2bf(o);
  }
}
__global__ __launch_bounds__(256)
void ln_concat768_k(const float* __restrict__ a, const float* __restrict__ bsrc,
                    const float* __restrict__ c, const float* __restrict__ g,
                    const float* __restrict__ bvec, unsigned short* __restrict__ out) {
  int rowi = blockIdx.x, t = threadIdx.x;
  long base = (long)rowi * 256 + t;
  float v0 = a[base], v1 = bsrc[base], v2 = c[base];
  float mean = blockSum256(v0 + v1 + v2) * (1.f / 768.f);
  float d0 = v0 - mean, d1 = v1 - mean, d2 = v2 - mean;
  float var = blockSum256(d0 * d0 + d1 * d1 + d2 * d2) * (1.f / 768.f);
  float rs = rsqrtf(var + 1e-5f);
  long ob = (long)rowi * 768;
  out[ob + t]       = f2bf(d0 * rs * g[t]       + bvec[t]);
  out[ob + 256 + t] = f2bf(d1 * rs * g[256 + t] + bvec[256 + t]);
  out[ob + 512 + t] = f2bf(d2 * rs * g[512 + t] + bvec[512 + t]);
}

// means over the set dimension; grid(D/256, B)
__global__ void mean_f32_k(const float* src, int S, int D, float* out) {
  int d = blockIdx.x * 256 + threadIdx.x, b = blockIdx.y;
  float s = 0.f;
  for (int i = 0; i < S; ++i) s += src[((long)b * S + i) * D + d];
  out[b * D + d] = s / (float)S;
}
__global__ void mean_bf16_k(const unsigned short* src, int S, int D, float* out) {
  int d = blockIdx.x * 256 + threadIdx.x, b = blockIdx.y;
  float s = 0.f;
  for (int i = 0; i < S; ++i) s += bf2f(src[((long)b * S + i) * D + d]);
  out[b * D + d] = s / (float)S;
}
// tiny y[b,n] = x[b,:K] @ W[K,N] (+bias) ; grid(N/256, B)
__global__ void vecmat_k(const float* x, int K, const float* W, int N,
                         const float* bias, float* y) {
  int n = blockIdx.x * 256 + threadIdx.x, b = blockIdx.y;
  float acc = bias ? bias[n] : 0.f;
  const float* xr = x + (long)b * K;
  for (int k = 0; k < K; ++k) acc += xr[k] * W[(long)k * N + n];
  y[(long)b * N + n] = acc;
}

// outputs: i_t | logit | pred_is_charged | e_t | n_t
__global__ void finalize_k(const float* it, const float* logitb, const int* mask,
                           const float* e, const float* nt, float* out) {
  int i = blockIdx.x * 256 + threadIdx.x;
  if (i >= 1279488) return;
  if (i < 294912)      out[i] = it[i];
  else if (i < 295680) out[i] = logitb[i - 294912];
  else if (i < 296448) { int j = i - 295680; int b = j / NE, ee = j % NE;
                         out[i] = mask[b * NV + ee] ? 1.f : 0.f; }
  else if (i < 493056) out[i] = e[i - 296448];
  else                 out[i] = nt[i - 493056];
}

// ---------------------------------------------------------------------------
// Host orchestration
// ---------------------------------------------------------------------------
extern "C" void kernel_launch(void* const* d_in, const int* in_sizes, int n_in,
                              void* d_out, int out_size, void* d_ws, size_t ws_size,
                              hipStream_t stream) {
  (void)in_sizes; (void)n_in; (void)out_size; (void)ws_size;

  // ---- inputs (setup_inputs dict order, params flattened depth-first) ----
  const float* f_inputs = (const float*)d_in[0];
  const int*   i_mask   = (const int*)d_in[1];
  const float* f_noise  = (const float*)d_in[2];
  const float* f_projw  = (const float*)d_in[3];
  const float* f_projb  = (const float*)d_in[4];
  const float* f_mu     = (const float*)d_in[5];
  const float* f_ls     = (const float*)d_in[6];
  const float* f_pew    = (const float*)d_in[7];
  const float* f_peb    = (const float*)d_in[8];
  const float* f_pnw    = (const float*)d_in[9];
  const float* f_pnb    = (const float*)d_in[10];
  const float* f_piw    = (const float*)d_in[11];
  const float* f_pib    = (const float*)d_in[12];
  const float* f_ocw    = (const float*)d_in[13];
  const float* f_ocb    = (const float*)d_in[14];
  const float* f_eiw1   = (const float*)d_in[15];
  const float* f_eib1   = (const float*)d_in[16];
  const float* f_eiw2   = (const float*)d_in[17];
  const float* f_eib2   = (const float*)d_in[18];
  const float* f_l1aw   = (const float*)d_in[19];
  const float* f_l1ab   = (const float*)d_in[20];
  const float* f_l1bw   = (const float*)d_in[21];
  const float* f_l1bb   = (const float*)d_in[22];
  const float* f_lng    = (const float*)d_in[23];
  const float* f_lnb    = (const float*)d_in[24];
  const float* f_l2aw   = (const float*)d_in[25];
  const float* f_l2ab   = (const float*)d_in[26];
  const float* f_l2bw   = (const float*)d_in[27];
  const float* f_l2bb   = (const float*)d_in[28];
  const float* f_npg    = (const float*)d_in[29];
  const float* f_npb    = (const float*)d_in[30];
  const float* f_nng    = (const float*)d_in[31];
  const float* f_nnb    = (const float*)d_in[32];
  const float* f_inw    = (const float*)d_in[33];
  const float* f_inb    = (const float*)d_in[34];
  const float* p_modw[2] = {(const float*)d_in[35], (const float*)d_in[45]};
  const float* p_modb[2] = {(const float*)d_in[36], (const float*)d_in[46]};
  const float* p_qkvw[2] = {(const float*)d_in[37], (const float*)d_in[47]};
  const float* p_qkvb[2] = {(const float*)d_in[38], (const float*)d_in[48]};
  const float* p_ow[2]   = {(const float*)d_in[39], (const float*)d_in[49]};
  const float* p_ob[2]   = {(const float*)d_in[40], (const float*)d_in[50]};
  const float* p_m1w[2]  = {(const float*)d_in[41], (const float*)d_in[51]};
  const float* p_m1b[2]  = {(const float*)d_in[42], (const float*)d_in[52]};
  const float* p_m2w[2]  = {(const float*)d_in[43], (const float*)d_in[53]};
  const float* p_m2b[2]  = {(const float*)d_in[44], (const float*)d_in[54]};
  const float* f_fng     = (const float*)d_in[55];
  const float* f_fnb     = (const float*)d_in[56];

  // ---- workspace bump allocator ----
  char* base = (char*)d_ws; size_t off = 0;
  auto AF = [&](size_t n) { float* p = (float*)(base + off); off += n * 4; return p; };
  auto AH = [&](size_t n) { unsigned short* p = (unsigned short*)(base + off); off += n * 2; return p; };

  // transposed bf16 weights (Wt[N][K])
  unsigned short* w_projT = AH(8192);
  unsigned short* w_pnT   = AH(65536);
  unsigned short* w_peT   = AH(65536);
  unsigned short* w_inT   = AH(131072);
  unsigned short* w_qkvT[2] = {AH(196608), AH(196608)};
  unsigned short* w_oT[2]   = {AH(65536),  AH(65536)};
  unsigned short* w_m1T[2]  = {AH(262144), AH(262144)};
  unsigned short* w_m2T[2]  = {AH(262144), AH(262144)};
  unsigned short* w_s1T = AH(196608);
  unsigned short* w_s2T = AH(65536);
  float* b1sum = AF(256); float* b2sum = AF(256);
  float* ctx = AF(2048);  float* sctx = AF(2048);
  float* cmod[2] = {AF(12288), AF(12288)};
  float* inpF = AF(786432);
  unsigned short* inpPad = AH(98304);
  float* ntF = AF(786432);
  unsigned short* ntB = AH(786432);    // [3072][256] row-major (A operand)
  unsigned short* ntTB = AH(786432);   // [8][256][384] K-contiguous (B operand)
  float* etF = AF(196608);
  unsigned short* cat2 = AH(393216);   // [768][512]: e_t | upd_e (bf16, A operand)
  unsigned short* etTB = AH(196608);   // [8][256][96] K-contiguous (B operand)
  float* itF = AF(294912); float* ilog = AF(294912);
  float* isum = AF(768); float* eind = AF(768); float* logitb = AF(768);
  unsigned short* imB = AH(294912); unsigned short* imTB = AH(294912);
  float* pnF = AF(786432); float* peF = AF(196608);
  float* xF = AF(196608);  unsigned short* hB = AH(196608);
  float* qkvF = AF(589824);
  unsigned short* qB = AH(196608); unsigned short* kBB = AH(196608);
  unsigned short* vTB = AH(196608);
  float* scF = AF(589824); unsigned short* aB = AH(589824);
  unsigned short* oB = AH(196608); unsigned short* ocB = AH(196608);
  unsigned short* hidB = AH(786432);
  float* updnF = AF(786432);
  unsigned short* hcatB = AH(2359296);
  float* xmF = AF(6144); float* mt1 = AF(2048);
  float* h2pre = AF(786432); float* h2F = AF(786432); unsigned short* h2B = AH(786432);
  float* h2m = AF(2048); float* mt2 = AF(2048);
  float* mloF = AF(786432);

  const float* FN = nullptr; float* FWN = nullptr; unsigned short* HN = nullptr;

  auto gemm = [&](const unsigned short* A, int lda, long sA,
                  const unsigned short* Bt, int ldbt, long sB,
                  float* C, int ldc, long sC,
                  unsigned short* Cb, int ldcb, long sCb,
                  int M, int N, int K, int batch,
                  const float* bias, const float* res, int ldres, long sres,
                  const float* gate, int gst, const float* mvec, int mst,
                  int rpb, float alpha, int act) {
    gemm_bf16<<<dim3(N / 32, M / 32, batch), 32, 0, stream>>>(
        A, lda, sA, Bt, ldbt, sB, C, ldc, sC, Cb, ldcb, sCb, K,
        bias, res, ldres, sres, gate, gst, mvec, mst, rpb, alpha, act);
  };
  auto cvtT = [&](const float* s, int K, int N, unsigned short* d) {
    int total = K * N;
    cvtT_bf16_k<<<(total + 255) / 256, 256, 0, stream>>>(s, K, N, d, total);
  };

  // ---- once-per-launch weight staging (bf16, transposed) ----
  padT_proj_k<<<32, 256, 0, stream>>>(f_projw, w_projT);
  pad_cols_bf16_k<<<384, 256, 0, stream>>>(f_inputs, 16, 32, 3072, inpPad);
  cvtT(f_pnw, 256, 256, w_pnT);
  cvtT(f_pew, 256, 256, w_peT);
  cvtT(f_inw, 512, 256, w_inT);
  for (int bl = 0; bl < 2; ++bl) {
    cvtT(p_qkvw[bl], 256, 768, w_qkvT[bl]);
    cvtT(p_ow[bl],   256, 256, w_oT[bl]);
    cvtT(p_m1w[bl],  256, 1024, w_m1T[bl]);
    cvtT(p_m2w[bl],  1024, 256, w_m2T[bl]);
  }
  add2T_bf16_k<<<768, 256, 0, stream>>>(f_l1aw, f_l1bw, 768, 256, w_s1T, 196608);
  add2T_bf16_k<<<256, 256, 0, stream>>>(f_l2aw, f_l2bw, 256, 256, w_s2T, 65536);
  addf_k<<<1, 256, 0, stream>>>(f_l1ab, f_l1bb, b1sum, 256);
  addf_k<<<1, 256, 0, stream>>>(f_l2ab, f_l2bb, b2sum, 256);

  // ---- prologue: inp, ctx, modulation, e/i init ----
  gemm(inpPad, 32, 0, w_projT, 32, 0, inpF, 256, 0, ntB, 256, 0,
       3072, 256, 32, 1, f_projb, FN, 0, 0, FN, 0, FN, 0, 0, 1.f, 0);
  hipMemcpyAsync(ntF, inpF, 786432 * sizeof(float), hipMemcpyDeviceToDevice, stream);
  bf16T_k<<<3072, 256, 0, stream>>>(inpF, 384, ntTB);
  mean_f32_k<<<dim3(1, 8), 256, 0, stream>>>(inpF, 384, 256, ctx);
  silu_k<<<8, 256, 0, stream>>>(ctx, sctx, 2048);
  for (int bl = 0; bl < 2; ++bl)
    vecmat_k<<<dim3(6, 8), 256, 0, stream>>>(sctx, 256, p_modw[bl], 1536, p_modb[bl], cmod[bl]);
  e_init_k<<<768, 256, 0, stream>>>(f_mu, f_ls, f_noise, etF);
  i_init_k<<<1152, 256, 0, stream>>>(i_mask, itF);
  pack_e_k<<<768, 256, 0, stream>>>(etF, cat2);

  // ---- T refine iterations ----
  for (int t = 0; t < Tit; ++t) {
    // pn / pe projections
    gemm(ntB, 256, 0, w_pnT, 256, 0, pnF, 256, 0, HN, 0, 0,
         3072, 256, 256, 1, f_pnb, FN, 0, 0, FN, 0, FN, 0, 0, 1.f, 0);
    gemm(cat2, 512, 0, w_peT, 256, 0, peF, 256, 0, HN, 0, 0,
         768, 256, 256, 1, f_peb, FN, 0, 0, FN, 0, FN, 0, 0, 1.f, 0);
    // fused OutCatLinear -> logits; softmax over edges + mask/teye
    outcat_k<<<dim3(48, 96, 8), 256, 0, stream>>>(pnF, peF, itF, f_piw, f_pib,
                                                  f_ocw, f_ocb, ilog);
    softmax_col_k<<<768, 128, 0, stream>>>(ilog, i_mask, itF);
    isum_k<<<96, 256, 0, stream>>>(itF, isum);
    edge_ind_k<<<768, 256, 0, stream>>>(etF, isum, f_eiw1, f_eib1, f_eiw2, f_eib2,
                                        i_mask, eind, logitb);
    im_k<<<1152, 256, 0, stream>>>(itF, eind, imB, imTB);
    // upd_e = im @ n_t  -> bf16 into cat2[:,256:512]
    gemm(imB, 384, (long)NE * 384, ntTB, 384, (long)256 * 384, FWN, 0, 0,
         cat2 + 256, 512, (long)NE * 512,
         96, 256, 384, 8, FN, FN, 0, 0, FN, 0, FN, 0, 0, 1.f, 0);
    // DiT input: x = [e_t, upd_e] @ in_w + in_b
    gemm(cat2, 512, 0, w_inT, 512, 0, xF, 256, 0, HN, 0, 0,
         768, 256, 512, 1, f_inb, FN, 0, 0, FN, 0, FN, 0, 0, 1.f, 0);

    for (int bl = 0; bl < 2; ++bl) {
      const float* c = cmod[bl];
      // attn branch
      ln_mod256_k<<<768, 256, 0, stream>>>(xF, c + 0, c + 256, 1536, 96, hB);
      gemm(hB, 256, 0, w_qkvT[bl], 256, 0, qkvF, 768, 0, HN, 0, 0,
           768, 768, 256, 1, p_qkvb[bl], FN, 0, 0, FN, 0, FN, 0, 0, 1.f, 0);
      qkv_split_k<<<2304, 256, 0, stream>>>(qkvF, qB, kBB, vTB);
      gemm(qB, 32, 3072, kBB, 32, 3072, scF, 96, 9216, HN, 0, 0,
           96, 96, 32, 64, FN, FN, 0, 0, FN, 0, FN, 0, 0, 0.17677669529663687f, 0);
      softmax_row_k<<<1536, 128, 0, stream>>>(scF, aB);
      gemm(aB, 96, 9216, vTB, 96, 3072, FWN, 0, 0, oB, 32, 3072,
           96, 32, 96, 64, FN, FN, 0, 0, FN, 0, FN, 0, 0, 1.f, 0);
      o_reorder_k<<<768, 256, 0, stream>>>(oB, ocB);
      gemm(ocB, 256, 0, w_oT[bl], 256, 0, xF, 256, 0, HN, 0, 0,
           768, 256, 256, 1, p_ob[bl], xF, 256, 0, c + 512, 1536, FN, 0, 96, 1.f, 0);
      // mlp branch
      ln_mod256_k<<<768, 256, 0, stream>>>(xF, c + 768, c + 1024, 1536, 96, hB);
      gemm(hB, 256, 0, w_m1T[bl], 256, 0, FWN, 0, 0, hidB, 1024, 0,
           768, 1024, 256, 1, p_m1b[bl], FN, 0, 0, FN, 0, FN, 0, 0, 1.f, 2);
      gemm(hidB, 1024, 0, w_m2T[bl], 1024, 0, xF, 256, 0, HN, 0, 0,
           768, 256, 1024, 1, p_m2b[bl], xF, 256, 0, c + 1280, 1536, FN, 0, 96, 1.f, 0);
    }
    // e_new = LN(x); f32 -> etF, bf16 -> cat2[:, :256], bf16^T -> etTB
    ln_affine256_k<<<768, 256, 0, stream>>>(xF, FN, f_fng, f_fnb, etF, cat2, 512,
                                            etTB, 96);
    // upd_n = im^T @ e_new
    gemm(imTB, 96, (long)NV * 96, etTB, 96, (long)256 * 96, updnF, 256, (long)NV * 256,
         HN, 0, 0, 384, 256, 96, 8, FN, FN, 0, 0, FN, 0, FN, 0, 0, 1.f, 0);
    // h_n = LN([inp, n_t, upd_n]) (bf16)
    ln_concat768_k<<<3072, 256, 0, stream>>>(inpF, ntF, updnF, f_npg, f_npb, hcatB);
    // deepset layer1: relu(h@(w1a+w1b) + b - mean@w1b) then LN
    mean_bf16_k<<<dim3(3, 8), 256, 0, stream>>>(hcatB, 384, 768, xmF);
    vecmat_k<<<dim3(1, 8), 256, 0, stream>>>(xmF, 768, f_l1bw, 256, FN, mt1);
    gemm(hcatB, 768, 0, w_s1T, 768, 0, h2pre, 256, 0, HN, 0, 0,
         3072, 256, 768, 1, b1sum, FN, 0, 0, FN, 0, mt1, 256, 384, 1.f, 1);
    ln_affine256_k<<<3072, 256, 0, stream>>>(h2pre, FN, f_lng, f_lnb, h2F, h2B, 256,
                                             HN, 1);
    // deepset layer2
    mean_f32_k<<<dim3(1, 8), 256, 0, stream>>>(h2F, 384, 256, h2m);
    vecmat_k<<<dim3(1, 8), 256, 0, stream>>>(h2m, 256, f_l2bw, 256, FN, mt2);
    gemm(h2B, 256, 0, w_s2T, 256, 0, mloF, 256, 0, HN, 0, 0,
         3072, 256, 256, 1, b2sum, FN, 0, 0, FN, 0, mt2, 256, 384, 1.f, 0);
    // n_new = LN(n_t + mlp_out); also refresh ntB / ntTB for next iteration
    ln_affine256_k<<<3072, 256, 0, stream>>>(ntF, mloF, f_nng, f_nnb, ntF, ntB, 256,
                                             ntTB, 384);
  }

  finalize_k<<<(1279488 + 255) / 256, 256, 0, stream>>>(itF, logitb, i_mask, etF, ntF,
                                                        (float*)d_out);
}
